// BaseModelCA_36902359007494
// MI455X (gfx1250) — matrix operand order repaired and checked by reading, not compiled, not run on hardware
//
#include <hip/hip_runtime.h>
#include <hip/hip_bf16.h>
#include <cstdint>

// ---------------- static problem config (matches reference) ----------------
#define NB      8          // batch N
#define LQ      1024
#define DMODEL  256
#define NHEAD   8
#define NLVL    4
#define NPNT    4
#define DHEAD   32         // DMODEL / NHEAD
#define DFFN    1024
#define TOTHW   12240      // 96*96 + 48*48 + 24*24 + 12*12

__device__ __constant__ int c_H[NLVL]   = {96, 48, 24, 12};
__device__ __constant__ int c_W[NLVL]   = {96, 48, 24, 12};
__device__ __constant__ int c_OFF[NLVL] = {0, 9216, 11520, 12096};

typedef __attribute__((ext_vector_type(16))) _Float16 v16h;
typedef __attribute__((ext_vector_type(8)))  float    v8f;

// ---------------------------------------------------------------------------
// Weight pre-pack: Bt[col][kpad] = f16(B[k][col]), zero-padded to KPAD.
// Makes each lane's WMMA B-fragment two contiguous 16B runs.
// ---------------------------------------------------------------------------
__global__ void __launch_bounds__(256)
pack_bt_kernel(const float* __restrict__ B, _Float16* __restrict__ Bt,
               int K, int KPAD, int Nc)
{
    const int idx = blockIdx.x * 256 + threadIdx.x;
    if (idx >= Nc * KPAD) return;
    const int col = idx / KPAD;
    const int k   = idx - col * KPAD;
    Bt[idx] = (_Float16)((k < K) ? B[(size_t)k * Nc + col] : 0.0f);
}

// ---------------------------------------------------------------------------
// WMMA GEMM (K compile-time):
//   Cout[row, colOff+col] = op( A[MxK] @ B[KxNc] + bias )
// One 16x16 tile per wave, 4 waves/block.
//   REMAP=false: grid = (ceil(Nc/64), M/16);     outRow = A row.
//   REMAP=true : grid = (ceil(Nc/64), HWl/16, NB);
//                aRow  = z*HWl   + y*16 + m   (A is (NB, HWl, K) flat)
//                outRow= z*TOTHW + lvlOff + y*16 + mo  (level-concat buffer)
// A: fp32 row-major, aligned float4 runs -> v_cvt_pk_f16_f32.
// B: pre-packed f16 transposed (Bt[col][KPAD]), two 16B runs per lane per step.
// ---------------------------------------------------------------------------
template <int K, bool REMAP>
__global__ void __launch_bounds__(128)
gemm_wmma_f16(const float* __restrict__ A, const _Float16* __restrict__ Bt,
              const float* __restrict__ bias, float* __restrict__ Cout,
              int Nc, int ldOut, int colOff, int relu,
              int HWl, int lvlOff)
{
    constexpr int KPAD = (K + 31) & ~31;

    const int wave = threadIdx.x >> 5;
    const int lane = threadIdx.x & 31;
    const int tc   = blockIdx.x * 4 + wave;       // column tile
    if (tc * 16 >= Nc) return;
    const int tr   = blockIdx.y;                  // row tile (within image if REMAP)

    const int m  = lane & 15;                     // A row / B,C column within tile
    const int hi = lane >> 4;                     // lane-half select

    const int aRow = (REMAP ? blockIdx.z * HWl : 0) + tr * 16 + m;
    const int bCol = tc * 16 + m;
    const int outRowBase = REMAP ? ((int)blockIdx.z * TOTHW + lvlOff + tr * 16)
                                 : tr * 16;

    // C/D layout: lane owns column bCol; VGPR r holds row (r + 8*hi).
    v8f c;
    const float bv = bias[bCol];
#pragma unroll
    for (int r = 0; r < 8; ++r) c[r] = bv;

    const float*    __restrict__ arow = A  + (size_t)aRow * (size_t)K;
    const _Float16* __restrict__ brow = Bt + (size_t)bCol * (size_t)KPAD;

#pragma unroll
    for (int k0 = 0; k0 < KPAD; k0 += 32) {
        // A 16x32 f16 lane layout: lanes 0-15 -> K 0..7 & 16..23,
        //                          lanes 16-31 -> K 8..15 & 24..31.
        v16h a;
        if constexpr (K == 16) {
            const float4 a0 = *(const float4*)(arow + hi * 8);
            const float4 a1 = *(const float4*)(arow + hi * 8 + 4);
            a[0] = (_Float16)a0.x; a[1] = (_Float16)a0.y;
            a[2] = (_Float16)a0.z; a[3] = (_Float16)a0.w;
            a[4] = (_Float16)a1.x; a[5] = (_Float16)a1.y;
            a[6] = (_Float16)a1.z; a[7] = (_Float16)a1.w;
#pragma unroll
            for (int i = 8; i < 16; ++i) a[i] = (_Float16)0.0f;
        } else {
            const float4 a0 = *(const float4*)(arow + k0 + hi * 8);
            const float4 a1 = *(const float4*)(arow + k0 + hi * 8 + 4);
            const float4 a2 = *(const float4*)(arow + k0 + 16 + hi * 8);
            const float4 a3 = *(const float4*)(arow + k0 + 16 + hi * 8 + 4);
            a[0]  = (_Float16)a0.x; a[1]  = (_Float16)a0.y;
            a[2]  = (_Float16)a0.z; a[3]  = (_Float16)a0.w;
            a[4]  = (_Float16)a1.x; a[5]  = (_Float16)a1.y;
            a[6]  = (_Float16)a1.z; a[7]  = (_Float16)a1.w;
            a[8]  = (_Float16)a2.x; a[9]  = (_Float16)a2.y;
            a[10] = (_Float16)a2.z; a[11] = (_Float16)a2.w;
            a[12] = (_Float16)a3.x; a[13] = (_Float16)a3.y;
            a[14] = (_Float16)a3.z; a[15] = (_Float16)a3.w;
        }

        // B 32x16 lane layout: lanes 0-15 hold K 0..7 & 16..23,
        //                      lanes 16-31 hold K 8..15 & 24..31 (matches A).
        v16h b;
#pragma unroll
        for (int s = 0; s < 8; ++s) {
            b[s]     = brow[k0 + hi * 8 + s];
            b[8 + s] = brow[k0 + 16 + hi * 8 + s];
        }

        c = __builtin_amdgcn_wmma_f32_16x16x32_f16(
                /*neg_a=*/false, a, /*neg_b=*/false, b,
                /*c_mod=*/(short)0, c, /*reuse_a=*/false, /*reuse_b=*/false);
    }

#pragma unroll
    for (int r = 0; r < 8; ++r) {
        const int mo     = r + hi * 8;
        const int outRow = outRowBase + mo;
        float vres = c[r];
        if (relu) vres = fmaxf(vres, 0.0f);
        Cout[(size_t)outRow * (size_t)ldOut + colOff + bCol] = vres;
    }
}

// ---------------------------------------------------------------------------
// loc & softmax kernel.  P row = [256 offsets | 128 attn logits].
// block = 128 threads = (h, l, p); grid = N*Lq.  Softmax over 16-lane groups.
// ---------------------------------------------------------------------------
__global__ void __launch_bounds__(128)
loc_attn_kernel(const float* __restrict__ P, const float* __restrict__ ref,
                float* __restrict__ loc, float* __restrict__ attnw)
{
    const int nq  = blockIdx.x;
    const int tid = threadIdx.x;
    const int h = tid >> 4, l = (tid >> 2) & 3, p = tid & 3;

    const float* prow = P + (size_t)nq * 384;
    const float offx = prow[((h * NLVL + l) * NPNT + p) * 2 + 0];
    const float offy = prow[((h * NLVL + l) * NPNT + p) * 2 + 1];
    const float refx = ref[((size_t)nq * NLVL + l) * 2 + 0];
    const float refy = ref[((size_t)nq * NLVL + l) * 2 + 1];

    const int pidx = ((nq * NHEAD + h) * NLVL + l) * NPNT + p;
    loc[(size_t)pidx * 2 + 0] = refx + offx / (float)c_W[l];
    loc[(size_t)pidx * 2 + 1] = refy + offy / (float)c_H[l];

    // softmax over the 16 (l,p) slots of this head (one 16-lane group)
    float logit = prow[256 + h * 16 + (l * NPNT + p)];
    float mx = logit;
#pragma unroll
    for (int d = 8; d >= 1; d >>= 1) mx = fmaxf(mx, __shfl_xor(mx, d, 16));
    const float e = __expf(logit - mx);
    float s = e;
#pragma unroll
    for (int d = 8; d >= 1; d >>= 1) s += __shfl_xor(s, d, 16);
    attnw[pidx] = e / s;
}

// ---------------------------------------------------------------------------
// Bilinear sampling + attention-weighted sum.
// block = 256 = 8 waves (one per head), lane = channel (DH=32 == wave32).
// grid = N*Lq.  4 coalesced 128B corner gathers per (l,p) point, L2-resident.
// ---------------------------------------------------------------------------
__global__ void __launch_bounds__(256)
msdeform_sample_kernel(const float* __restrict__ val, const float* __restrict__ loc,
                       const float* __restrict__ attnw, float* __restrict__ ms_out)
{
    const int nq = blockIdx.x;
    const int n  = nq >> 10;
    const int h  = threadIdx.x >> 5;
    const int ch = threadIdx.x & 31;

    float acc = 0.0f;
#pragma unroll
    for (int l = 0; l < NLVL; ++l) {
        const int H = c_H[l], W = c_W[l];
        const size_t lvlBase = ((size_t)n * TOTHW + c_OFF[l]) * DMODEL;
#pragma unroll
        for (int p = 0; p < NPNT; ++p) {
            const int pidx = ((nq * NHEAD + h) * NLVL + l) * NPNT + p;
            const float lx = loc[(size_t)pidx * 2 + 0];
            const float ly = loc[(size_t)pidx * 2 + 1];
            const float aw = attnw[pidx];

            const float x = lx * (float)W - 0.5f;
            const float y = ly * (float)H - 0.5f;
            const float x0f = floorf(x), y0f = floorf(y);
            const int   x0 = (int)x0f,  y0 = (int)y0f;
            const float wx = x - x0f,   wy = y - y0f;

            float s = 0.0f;
#pragma unroll
            for (int dy = 0; dy < 2; ++dy) {
#pragma unroll
                for (int dx = 0; dx < 2; ++dx) {
                    const int yi = y0 + dy, xi = x0 + dx;
                    const float wgt = (dy ? wy : 1.0f - wy) * (dx ? wx : 1.0f - wx);
                    if (yi >= 0 && yi < H && xi >= 0 && xi < W) {
                        const size_t idx = lvlBase + ((size_t)yi * W + xi) * DMODEL
                                         + h * DHEAD + ch;
                        s += wgt * val[idx];
                    }
                }
            }
            acc += aw * s;
        }
    }
    ms_out[(size_t)nq * DMODEL + h * DHEAD + ch] = acc;
}

// ---------------------------------------------------------------------------
// out = LayerNorm(x + y) * g + b    (row = one block of 256 threads, D=256)
// ---------------------------------------------------------------------------
__global__ void __launch_bounds__(256)
add_ln_kernel(const float* __restrict__ x, const float* __restrict__ y,
              const float* __restrict__ g, const float* __restrict__ b,
              float* __restrict__ out)
{
    __shared__ float red[256];
    const int row = blockIdx.x, tid = threadIdx.x;
    const float v = x[(size_t)row * DMODEL + tid] + y[(size_t)row * DMODEL + tid];

    red[tid] = v; __syncthreads();
#pragma unroll
    for (int s = 128; s > 0; s >>= 1) {
        if (tid < s) red[tid] += red[tid + s];
        __syncthreads();
    }
    const float mean = red[0] * (1.0f / DMODEL);
    __syncthreads();

    const float dv = v - mean;
    red[tid] = dv * dv; __syncthreads();
#pragma unroll
    for (int s = 128; s > 0; s >>= 1) {
        if (tid < s) red[tid] += red[tid + s];
        __syncthreads();
    }
    const float var = red[0] * (1.0f / DMODEL);

    out[(size_t)row * DMODEL + tid] = dv * rsqrtf(var + 1e-5f) * g[tid] + b[tid];
}

// ---------------------------------------------------------------------------
extern "C" void kernel_launch(void* const* d_in, const int* in_sizes, int n_in,
                              void* d_out, int out_size, void* d_ws, size_t ws_size,
                              hipStream_t stream)
{
    const float* tgt   = (const float*)d_in[0];
    const float* ref   = (const float*)d_in[1];
    const float* src[4] = {(const float*)d_in[2], (const float*)d_in[3],
                           (const float*)d_in[4], (const float*)d_in[5]};
    const float* Wv0  = (const float*)d_in[6];  const float* bv0 = (const float*)d_in[7];
    const float* Wv1  = (const float*)d_in[8];  const float* bv1 = (const float*)d_in[9];
    const float* Woff = (const float*)d_in[10]; const float* boff = (const float*)d_in[11];
    const float* Wattn= (const float*)d_in[12]; const float* battn= (const float*)d_in[13];
    const float* Wout = (const float*)d_in[14]; const float* bout = (const float*)d_in[15];
    const float* g1   = (const float*)d_in[16]; const float* be1  = (const float*)d_in[17];
    const float* W1   = (const float*)d_in[18]; const float* b1   = (const float*)d_in[19];
    const float* W2   = (const float*)d_in[20]; const float* b2   = (const float*)d_in[21];
    const float* g3   = (const float*)d_in[22]; const float* be3  = (const float*)d_in[23];
    float* out = (float*)d_out;

    // ---- workspace carve-up (256B-aligned offsets) ----
    const int NQ = NB * LQ;                       // 8192
    uint8_t* ws = (uint8_t*)d_ws;
    size_t o = 0;
    auto carve = [&](size_t bytes) -> void* { void* p = (void*)(ws + o);
                                              o += (bytes + 255) & ~(size_t)255;
                                              return p; };
    float* val   = (float*)carve((size_t)NB * TOTHW * DMODEL * 4);  // ~100 MB
    float* P     = (float*)carve((size_t)NQ * 384 * 4);
    float* loc   = (float*)carve((size_t)NQ * NHEAD * NLVL * NPNT * 2 * 4);
    float* attnw = (float*)carve((size_t)NQ * NHEAD * NLVL * NPNT * 4);
    float* ms    = (float*)carve((size_t)NQ * DMODEL * 4);
    float* msp   = (float*)carve((size_t)NQ * DMODEL * 4);
    float* t     = (float*)carve((size_t)NQ * DMODEL * 4);
    float* hbuf  = (float*)carve((size_t)NQ * DFFN * 4);
    float* fbuf  = (float*)carve((size_t)NQ * DMODEL * 4);
    // packed f16 transposed weights (Bt[col][KPAD])
    _Float16* BtV0   = (_Float16*)carve((size_t)DMODEL * 32 * 2);   // K=16 ->32
    _Float16* BtV1   = (_Float16*)carve((size_t)DMODEL * 32 * 2);   // K=32
    _Float16* BtOff  = (_Float16*)carve((size_t)256 * DMODEL * 2);  // Nc=256,K=256
    _Float16* BtAttn = (_Float16*)carve((size_t)128 * DMODEL * 2);  // Nc=128,K=256
    _Float16* BtOut  = (_Float16*)carve((size_t)256 * DMODEL * 2);  // Nc=256,K=256
    _Float16* BtW1   = (_Float16*)carve((size_t)DFFN * DMODEL * 2); // Nc=1024,K=256
    _Float16* BtW2   = (_Float16*)carve((size_t)DMODEL * DFFN * 2); // Nc=256,K=1024
    (void)ws_size; (void)in_sizes; (void)n_in; (void)out_size;

    auto packGrid = [](int n) { return (n + 255) / 256; };
    // 0) pack all weights to f16^T once per launch (deterministic, ~1.4 MB)
    pack_bt_kernel<<<packGrid(DMODEL * 32),   256, 0, stream>>>(Wv0,  BtV0,   16,   32, DMODEL);
    pack_bt_kernel<<<packGrid(DMODEL * 32),   256, 0, stream>>>(Wv1,  BtV1,   32,   32, DMODEL);
    pack_bt_kernel<<<packGrid(256 * DMODEL),  256, 0, stream>>>(Woff, BtOff,  256,  256, 256);
    pack_bt_kernel<<<packGrid(128 * DMODEL),  256, 0, stream>>>(Wattn,BtAttn, 256,  256, 128);
    pack_bt_kernel<<<packGrid(256 * DMODEL),  256, 0, stream>>>(Wout, BtOut,  256,  256, 256);
    pack_bt_kernel<<<packGrid(DFFN * DMODEL), 256, 0, stream>>>(W1,   BtW1,   256,  256, DFFN);
    pack_bt_kernel<<<packGrid(DMODEL * DFFN), 256, 0, stream>>>(W2,   BtW2,   1024, 1024, DMODEL);

    const int HWs[4]  = {9216, 2304, 576, 144};
    const int OFFs[4] = {0, 9216, 11520, 12096};

    // 1) per-level value projection: src_l (NB, HW, Cin) @ Wv (Cin x 256) -> val
    //    grid = (colTiles/4, HW/16, NB): no division in the store path.
    {
        const _Float16* Bts[4] = {BtV0, BtV1, BtV0, BtV1};
        const float*    bvs[4] = {bv0,  bv1,  bv0,  bv1};
        for (int l = 0; l < 4; ++l) {
            const dim3 grid(4, HWs[l] / 16, NB);
            if (l == 0 || l == 2)
                gemm_wmma_f16<16, true><<<grid, 128, 0, stream>>>(
                    src[l], Bts[l], bvs[l], val, DMODEL, DMODEL, 0, 0,
                    HWs[l], OFFs[l]);
            else
                gemm_wmma_f16<32, true><<<grid, 128, 0, stream>>>(
                    src[l], Bts[l], bvs[l], val, DMODEL, DMODEL, 0, 0,
                    HWs[l], OFFs[l]);
        }
    }

    // 2) P = tgt @ [Woff | Wattn] + [boff | battn]
    gemm_wmma_f16<256, false><<<dim3(4, NQ / 16), 128, 0, stream>>>(
        tgt, BtOff, boff, P, 256, 384, 0, 0, 0, 0);
    gemm_wmma_f16<256, false><<<dim3(2, NQ / 16), 128, 0, stream>>>(
        tgt, BtAttn, battn, P, 128, 384, 256, 0, 0, 0);

    // 3) sampling locations + softmax attention weights
    loc_attn_kernel<<<NQ, 128, 0, stream>>>(P, ref, loc, attnw);

    // 4) bilinear gather + weighted sum -> ms (NQ x 256)
    msdeform_sample_kernel<<<NQ, 256, 0, stream>>>(val, loc, attnw, ms);

    // 5) output projection
    gemm_wmma_f16<256, false><<<dim3(4, NQ / 16), 128, 0, stream>>>(
        ms, BtOut, bout, msp, 256, 256, 0, 0, 0, 0);

    // 6) t = LN(tgt + msp)
    add_ln_kernel<<<NQ, 256, 0, stream>>>(tgt, msp, g1, be1, t);

    // 7) FFN up (relu) and down
    gemm_wmma_f16<256, false><<<dim3(16, NQ / 16), 128, 0, stream>>>(
        t, BtW1, b1, hbuf, DFFN, DFFN, 0, /*relu=*/1, 0, 0);
    gemm_wmma_f16<1024, false><<<dim3(4, NQ / 16), 128, 0, stream>>>(
        hbuf, BtW2, b2, fbuf, 256, 256, 0, 0, 0, 0);

    // 8) out = LN(t + fbuf)
    add_ln_kernel<<<NQ, 256, 0, stream>>>(t, fbuf, g3, be3, out);
}
